// MambaVisionMixer_16621523436201
// MI455X (gfx1250) — matrix operand order repaired and checked by reading, not compiled, not run on hardware
//
#include <hip/hip_runtime.h>
#include <hip/hip_bf16.h>

// Problem constants (compile-time, from the reference)
#define BB     4
#define LL     2048
#define DMODEL 768
#define NSTATE 16
#define DIN_   1536
#define D2_    768
#define DTR_   48
#define KPAD   64                 // dt_proj K=48 zero-padded to 64
#define MTOT   (BB * LL)          // 8192 rows for every activation GEMM

typedef __attribute__((ext_vector_type(16))) __bf16 v16bf;
typedef __attribute__((ext_vector_type(8)))  __bf16 v8bf;
typedef __attribute__((ext_vector_type(8)))  float  v8f;

__device__ __forceinline__ __bf16 f2bf(float f) {
  union { float f; unsigned u; } v; v.f = f;
  unsigned r = (v.u + 0x7FFFu + ((v.u >> 16) & 1u)) >> 16;   // RNE
  union { unsigned short s; __bf16 b; } o; o.s = (unsigned short)r;
  return o.b;
}

// ---------------------------------------------------------------------------
// bf16 WMMA GEMM:  C[m,n] = epilogue( sum_k A[m,k] * Wt[k,n] )
//   A : [M, lda] bf16 row-major.   Wt: [K, N] bf16 row-major (pre-transposed).
//   K % 32 == 0, M % 16 == 0, N % 16 == 0  -> no per-element guards.
//   One wave -> NT adjacent 16x16 tiles; A fragment reused NT times.
//   Software-pipelined: next k-step's fragments are loaded into a separate
//   register buffer before the current WMMA chain, keeping a full step of
//   loads in flight behind the WMMAs.
//   OUTMODE 0: fp32 [M,N].  OUTMODE 1: fp32 [B, N, L].
//   EPI 1: softplus(acc + 2*bias[n])   (dt_proj double-bias quirk).
// ---------------------------------------------------------------------------
template <int NT, int OUTMODE, int EPI>
__global__ void gemm_wmma_bf16(const __bf16* __restrict__ A, int lda,
                               const __bf16* __restrict__ Wt,
                               const float* __restrict__ bias,
                               float* __restrict__ C,
                               int M, int N, int K) {
  const int wave = threadIdx.x >> 5;
  const int lane = threadIdx.x & 31;
  const int nBase = (blockIdx.x * 8 + wave) * (NT * 16);
  const int mBase = blockIdx.y * 16;
  if (nBase >= N) return;                      // whole-wave guard (N=80 case)

  const int halfLane = lane & 15;
  const int hi       = lane >> 4;              // lane group 0 / 1
  const int rowA     = mBase + halfLane;       // A-matrix: lane indexes M
  const int kOffA    = hi * 8;                 // lanes 16-31: K+8 / K+24 groups

  v8f acc[NT];
#pragma unroll
  for (int t = 0; t < NT; ++t) acc[t] = v8f{};

  const __bf16* aptr = A + (size_t)rowA * lda + kOffA;
  const __bf16* bptr = Wt + (size_t)kOffA * N + nBase + halfLane;     // lane indexes N

  auto mma_chain = [&](v8bf a0, v8bf a1, v8bf (&b0)[NT], v8bf (&b1)[NT]) {
    v16bf a;
#pragma unroll
    for (int i = 0; i < 8; ++i) { a[i] = a0[i]; a[i + 8] = a1[i]; }
#pragma unroll
    for (int t = 0; t < NT; ++t) {
      v16bf b;
#pragma unroll
      for (int i = 0; i < 8; ++i) { b[i] = b0[t][i]; b[i + 8] = b1[t][i]; }
      acc[t] = __builtin_amdgcn_wmma_f32_16x16x32_bf16(
          false, a, false, b, (short)0, acc[t], false, false);
    }
  };

  // ---- prologue: load k-step 0
  v8bf a0c = *(const v8bf*)(aptr);             // K = off    .. off+7
  v8bf a1c = *(const v8bf*)(aptr + 16);        // K = off+16 .. off+23
  v8bf b0c[NT], b1c[NT];
#pragma unroll
  for (int t = 0; t < NT; ++t) {
    for (int i = 0; i < 8; ++i) { b0c[t][i] = bptr[(size_t)i * N + t * 16]; b1c[t][i] = bptr[(size_t)(i + 16) * N + t * 16]; }
  }

  // ---- pipelined steps: issue loads for s+1, then WMMA chain for s
  const int nsteps = K >> 5;
  for (int s = 0; s < nsteps - 1; ++s) {
    aptr += 32;
    bptr += (size_t)32 * N;
    v8bf a0n = *(const v8bf*)(aptr);
    v8bf a1n = *(const v8bf*)(aptr + 16);
    v8bf b0n[NT], b1n[NT];
#pragma unroll
    for (int t = 0; t < NT; ++t) {
      for (int i = 0; i < 8; ++i) { b0n[t][i] = bptr[(size_t)i * N + t * 16]; b1n[t][i] = bptr[(size_t)(i + 16) * N + t * 16]; }
    }

    mma_chain(a0c, a1c, b0c, b1c);

    a0c = a0n; a1c = a1n;
#pragma unroll
    for (int t = 0; t < NT; ++t) { b0c[t] = b0n[t]; b1c[t] = b1n[t]; }
  }
  // ---- drain
  mma_chain(a0c, a1c, b0c, b1c);

  // C/D layout: VGPR v -> M = mBase + v (+8 for lanes 16-31), N = lane&15
#pragma unroll
  for (int t = 0; t < NT; ++t) {
#pragma unroll
    for (int v = 0; v < 8; ++v) {
      int m = mBase + v + hi * 8;
      int n = nBase + t * 16 + halfLane;
      float val = acc[t][v];
      if (EPI == 1) {
        float x = val + 2.0f * bias[n];              // double-bias quirk
        val = (x > 20.0f) ? x : log1pf(__expf(x));   // softplus
      }
      if (OUTMODE == 0) {
        C[(size_t)m * N + n] = val;
      } else {
        int bIdx = m >> 11, l = m & (LL - 1);        // L = 2048
        C[((size_t)bIdx * N + n) * LL + l] = val;
      }
    }
  }
}

// ---------------------------------------------------------------------------
// Prep: fp32 -> bf16 copy
__global__ void cvt_bf16_kernel(const float* __restrict__ x,
                                __bf16* __restrict__ y, int n) {
  int i = blockIdx.x * blockDim.x + threadIdx.x;
  if (i < n) y[i] = f2bf(x[i]);
}

// Prep: W [N,K] fp32  ->  Wt [KP,N] bf16, rows K..KP-1 zeroed
__global__ void wt_kernel(const float* __restrict__ W, __bf16* __restrict__ Wt,
                          int N, int K, int KP) {
  int i = blockIdx.x * blockDim.x + threadIdx.x;
  if (i >= KP * N) return;
  int n = i % N, k = i / N;
  Wt[i] = (k < K) ? f2bf(W[(size_t)n * K + k]) : f2bf(0.0f);
}

// Prep: xdbl fp32 [M,80] cols 0..47 -> dtr_h bf16 [M,64], cols 48..63 zero
__global__ void dtr_pad_kernel(const float* __restrict__ xdbl,
                               __bf16* __restrict__ dtr_h) {
  int i = blockIdx.x * blockDim.x + threadIdx.x;
  if (i >= MTOT * KPAD) return;
  int c = i % KPAD, m = i / KPAD;
  dtr_h[i] = (c < DTR_) ? f2bf(xdbl[(size_t)m * 80 + c]) : f2bf(0.0f);
}

// ---------------------------------------------------------------------------
// Depthwise 'same' conv (K=4, pad L1/R2) + SiLU on xz [B, DIN, L].
//   32(C) x 32(L) tile per block; results staged in LDS (stride 33 ->
//   conflict-free) so BOTH the [B,C,L] reads and the [B,L,C] writes coalesce.
//   c in [0,768):    x path -> u fp32 + u_h bf16   [B,L,D2]
//   c in [768,1536): z path -> ycat_h bf16 [B,L,DIN] cols 768..1535
// ---------------------------------------------------------------------------
__global__ void conv_silu_kernel(const float* __restrict__ xz,
                                 const float* __restrict__ wx,
                                 const float* __restrict__ wz,
                                 float* __restrict__ u,
                                 __bf16* __restrict__ u_h,
                                 __bf16* __restrict__ ycat_h) {
  __shared__ float sm[32][33];
  const int l0 = blockIdx.x * 32;
  const int c0 = blockIdx.y * 32;
  const int b  = blockIdx.z;
  const int g    = threadIdx.x >> 5;
  const int lane = threadIdx.x & 31;

  // phase 1: conv + SiLU, reads coalesced along L
#pragma unroll
  for (int j = 0; j < 4; ++j) {
    int cl = g * 4 + j;
    int c  = c0 + cl;
    int l  = l0 + lane;
    const float* row = xz + ((size_t)b * DIN_ + c) * LL;
    const float* w = (c < D2_) ? (wx + c * 4) : (wz + (c - D2_) * 4);
    float s = 0.0f;
#pragma unroll
    for (int k = 0; k < 4; ++k) {
      int p = l + k - 1;                        // left pad 1, right pad 2
      if (p >= 0 && p < LL) s += row[p] * w[k];
    }
    sm[cl][lane] = s * (1.0f / (1.0f + __expf(-s)));   // SiLU
  }
  __syncthreads();

  // phase 2: transposed write-out, coalesced along C
#pragma unroll
  for (int j = 0; j < 4; ++j) {
    int ll = g * 4 + j;
    int l  = l0 + ll;
    int c  = c0 + lane;
    float y = sm[lane][ll];
    if (c0 < D2_) {                              // block-uniform branch
      size_t o = ((size_t)b * LL + l) * D2_ + c;
      u[o] = y;
      u_h[o] = f2bf(y);
    } else {
      ycat_h[((size_t)b * LL + l) * DIN_ + c] = f2bf(y);
    }
  }
}

// ---------------------------------------------------------------------------
// Selective scan: one thread per (b, d); sequential over L.
//   h[n] = exp(delta*A[d,n])*h[n] + delta*B[t,n]*u ;  y = sum_n h[n]*C[t,n]
//   B/C rows staged through LDS per block.  Writes bf16 into ycat cols 0..767.
// ---------------------------------------------------------------------------
#define TCH 64
__global__ void scan_kernel(const float* __restrict__ delta,   // [B, L, D2]
                            const float* __restrict__ u,       // [B, L, D2]
                            const float* __restrict__ xdbl,    // [B, L, 80]
                            const float* __restrict__ A_log,   // [D2, 16]
                            const float* __restrict__ Dp,      // [D2]
                            __bf16* __restrict__ ycat_h) {     // [B, L, DIN]
  const int d = blockIdx.x * blockDim.x + threadIdx.x;
  const int b = blockIdx.y;

  float Arow[NSTATE];
  float Dv = 0.0f;
  if (d < D2_) {
#pragma unroll
    for (int n = 0; n < NSTATE; ++n) Arow[n] = -__expf(A_log[d * NSTATE + n]);
    Dv = Dp[d];
  }
  float h[NSTATE];
#pragma unroll
  for (int n = 0; n < NSTATE; ++n) h[n] = 0.0f;

  __shared__ float sB[TCH][NSTATE];
  __shared__ float sC[TCH][NSTATE];

  for (int t0 = 0; t0 < LL; t0 += TCH) {
    __syncthreads();
    for (int i = threadIdx.x; i < TCH * NSTATE; i += blockDim.x) {
      int tt = i / NSTATE, n = i % NSTATE;
      size_t base = ((size_t)b * LL + t0 + tt) * 80;
      sB[tt][n] = xdbl[base + DTR_ + n];             // cols 48..63
      sC[tt][n] = xdbl[base + DTR_ + NSTATE + n];    // cols 64..79
    }
    __syncthreads();
    if (d >= D2_) continue;
    for (int tt = 0; tt < TCH; ++tt) {
      size_t off = ((size_t)b * LL + t0 + tt) * D2_ + d;
      float dl = delta[off];
      float uv = u[off];
      float du = dl * uv;
      float y = 0.0f;
#pragma unroll
      for (int n = 0; n < NSTATE; ++n) {
        h[n] = __expf(dl * Arow[n]) * h[n] + du * sB[tt][n];
        y += h[n] * sC[tt][n];
      }
      ycat_h[((size_t)b * LL + (t0 + tt)) * DIN_ + d] = f2bf(y + uv * Dv);
    }
  }
}

// ---------------------------------------------------------------------------
extern "C" void kernel_launch(void* const* d_in, const int* in_sizes, int n_in,
                              void* d_out, int out_size, void* d_ws, size_t ws_size,
                              hipStream_t stream) {
  const float* hidden     = (const float*)d_in[0];  // [B, L, D]
  const float* in_proj_w  = (const float*)d_in[1];  // [1536, 768]
  const float* x_proj_w   = (const float*)d_in[2];  // [80, 768]
  const float* dt_proj_w  = (const float*)d_in[3];  // [768, 48]
  const float* dt_proj_b  = (const float*)d_in[4];  // [768]
  const float* A_log      = (const float*)d_in[5];  // [768, 16]
  const float* D_param    = (const float*)d_in[6];  // [768]
  const float* out_proj_w = (const float*)d_in[7];  // [768, 1536]
  const float* conv_x_w   = (const float*)d_in[8];  // [768,1,4]
  const float* conv_z_w   = (const float*)d_in[9];  // [768,1,4]
  float* out = (float*)d_out;                       // [B, L, 768]

  // Workspace layout
  char* ws = (char*)d_ws;
  float*  xz       = (float*)ws;  ws += (size_t)BB * DIN_ * LL * 4;   // 50.3 MB
  float*  u        = (float*)ws;  ws += (size_t)BB * LL * D2_ * 4;    // 25.2 MB
  float*  xdbl     = (float*)ws;  ws += (size_t)MTOT * 80 * 4;        //  2.6 MB
  float*  delta    = (float*)ws;  ws += (size_t)MTOT * D2_ * 4;       // 25.2 MB
  __bf16* hid_h    = (__bf16*)ws; ws += (size_t)MTOT * DMODEL * 2;    // 12.6 MB
  __bf16* u_h      = (__bf16*)ws; ws += (size_t)MTOT * D2_ * 2;       // 12.6 MB
  __bf16* ycat_h   = (__bf16*)ws; ws += (size_t)MTOT * DIN_ * 2;      // 25.2 MB
  __bf16* dtr_h    = (__bf16*)ws; ws += (size_t)MTOT * KPAD * 2;      //  1.0 MB
  __bf16* in_wt    = (__bf16*)ws; ws += (size_t)DMODEL * DIN_ * 2;    //  2.4 MB
  __bf16* x_wt     = (__bf16*)ws; ws += (size_t)DMODEL * 80 * 2;      //  0.1 MB
  __bf16* dt_wt    = (__bf16*)ws; ws += (size_t)KPAD * D2_ * 2;       //  0.1 MB
  __bf16* out_wt   = (__bf16*)ws;                                     //  2.4 MB

  dim3 blk(256);
  auto g1 = [](int n) { return dim3((n + 255) / 256); };

  // --- Prep: bf16 conversions + weight transposes (small, once per call)
  cvt_bf16_kernel<<<g1(MTOT * DMODEL), blk, 0, stream>>>(hidden, hid_h, MTOT * DMODEL);
  wt_kernel<<<g1(DMODEL * DIN_), blk, 0, stream>>>(in_proj_w,  in_wt,  DIN_,  DMODEL, DMODEL);
  wt_kernel<<<g1(DMODEL * 80),   blk, 0, stream>>>(x_proj_w,   x_wt,   80,    DMODEL, DMODEL);
  wt_kernel<<<g1(KPAD * D2_),    blk, 0, stream>>>(dt_proj_w,  dt_wt,  D2_,   DTR_,   KPAD);
  wt_kernel<<<g1(DIN_ * DMODEL), blk, 0, stream>>>(out_proj_w, out_wt, DMODEL, DIN_,  DIN_);

  // 1) in_proj: [8192,768]x[768,1536] -> xz channel-major [B,1536,L]
  //    NT=4: block covers 512 cols; grid.x = 1536/512 = 3
  gemm_wmma_bf16<4, 1, 0><<<dim3(3, MTOT / 16), blk, 0, stream>>>(
      hid_h, DMODEL, in_wt, nullptr, xz, MTOT, DIN_, DMODEL);

  // 2) depthwise conv + SiLU: x -> u/u_h, z -> ycat_h[:, 768:]
  conv_silu_kernel<<<dim3(LL / 32, DIN_ / 32, BB), blk, 0, stream>>>(
      xz, conv_x_w, conv_z_w, u, u_h, ycat_h);

  // 3) x_proj: [8192,768]x[768,80] -> xdbl fp32 (NT=1, guard handles 5 tiles)
  gemm_wmma_bf16<1, 0, 0><<<dim3(1, MTOT / 16), blk, 0, stream>>>(
      u_h, D2_, x_wt, nullptr, xdbl, MTOT, 80, DMODEL);

  // 4a) pad dtr cols to K=64 bf16
  dtr_pad_kernel<<<g1(MTOT * KPAD), blk, 0, stream>>>(xdbl, dtr_h);

  // 4b) dt_proj + softplus(acc + 2b): [8192,64]x[64,768] -> delta
  //     NT=2: block covers 256 cols; grid.x = 768/256 = 3
  gemm_wmma_bf16<2, 0, 1><<<dim3(3, MTOT / 16), blk, 0, stream>>>(
      dtr_h, KPAD, dt_wt, dt_proj_b, delta, MTOT, D2_, KPAD);

  // 5) selective scan -> ycat_h[:, 0:768]
  scan_kernel<<<dim3((D2_ + 255) / 256, BB), blk, 0, stream>>>(
      delta, u, xdbl, A_log, D_param, ycat_h);

  // 6) out_proj: [8192,1536]x[1536,768] -> out fp32
  gemm_wmma_bf16<2, 0, 0><<<dim3(3, MTOT / 16), blk, 0, stream>>>(
      ycat_h, DIN_, out_wt, nullptr, out, MTOT, D2_, DIN_);
}
